// PFE_28037546508649
// MI455X (gfx1250) — matrix operand order repaired and checked by reading, not compiled, not run on hardware
//
#include <hip/hip_runtime.h>

// ---------------------------------------------------------------------------
// PointPillars PFE for MI455X (gfx1250, wave32, WMMA)
//   kernel 1: zero-fill the 219MB channel-major output map (float4 stores)
//   kernel 2: one wave per pillar:
//     - lane t loads point t (5x float4), wave shfl-xor reduce -> xyz mean
//     - 26ch features -> f16 -> LDS (4x ds_store_b128 per lane)
//     - A fragments from LDS per 16-bit A 16x32 layout, B (weights) fragments
//       resident in registers, 8x v_wmma_f32_16x16x32_f16 (f32 accum)
//     - BN+ReLU+max on accumulators, shfl_xor(16) cross-half max
//     - scatter 64 channel outputs (2 stores/lane)
// ---------------------------------------------------------------------------

typedef __attribute__((ext_vector_type(16))) _Float16 v16h;
typedef __attribute__((ext_vector_type(8)))  _Float16 v8h;
typedef __attribute__((ext_vector_type(8)))  float    v8f;

#define T_PTS   32
#define RAW_CH  20
#define C_IN    26
#define C_OUTN  64
#define NXC     432
#define NYC     496
#define NYX     (432 * 496)
#define BN_EPS  1e-3f
#define WAVES_PER_BLOCK 4

__global__ void pfe_zero_kernel(float4* __restrict__ out, int n4) {
    int i = blockIdx.x * blockDim.x + threadIdx.x;
    const int stride = gridDim.x * blockDim.x;
    const float4 z = make_float4(0.f, 0.f, 0.f, 0.f);
    for (; i < n4; i += stride) out[i] = z;
}

__global__ __launch_bounds__(WAVES_PER_BLOCK * 32)
void pfe_wmma_kernel(const float* __restrict__ pillars,
                     const float* __restrict__ W,
                     const float* __restrict__ gamma,
                     const float* __restrict__ beta,
                     const float* __restrict__ run_mean,
                     const float* __restrict__ run_var,
                     const int*   __restrict__ coords,
                     const int*   __restrict__ num_points,
                     float*       __restrict__ out,
                     int nP) {
    __shared__ _Float16 lds_feats[WAVES_PER_BLOCK][T_PTS][32];  // 8 KB

    const int lane       = threadIdx.x & 31;
    const int wave       = threadIdx.x >> 5;
    const int lcol       = lane & 15;
    const int halfSel    = lane >> 4;                // 0: lanes 0-15, 1: lanes 16-31
    const int waveGlobal = blockIdx.x * WAVES_PER_BLOCK + wave;
    const int nWaves     = gridDim.x * WAVES_PER_BLOCK;

    // ---- B fragments: W is (C_OUT=64 x C_IN=26) row-major; B[k][n] = W[n][k].
    // B 32x16 in contiguous K order (this code's choice, same as A): lanes 0-15 hold K=0..15 (2/VGPR),
    // lanes 16-31 hold K=16..31; lane%16 selects column N. K padded 26->32.
    v16h bfrag[4];
#pragma unroll
    for (int nt = 0; nt < 4; ++nt) {
        const int o = nt * 16 + lcol;
#pragma unroll
        for (int i = 0; i < 16; ++i) {
            const int k = i + halfSel * 16;
            const float w = (k < C_IN) ? W[o * C_IN + k] : 0.0f;
            bfrag[nt][i] = (_Float16)w;
        }
    }

    // ---- folded BN constants for this lane's column (per N-tile)
    float bnScale[4], bnBias[4];
#pragma unroll
    for (int nt = 0; nt < 4; ++nt) {
        const int o = nt * 16 + lcol;
        const float s = gamma[o] * rsqrtf(run_var[o] + BN_EPS);
        bnScale[nt] = s;
        bnBias[nt]  = beta[o] - run_mean[o] * s;
    }

    for (int p = waveGlobal; p < nP; p += nWaves) {
        // prefetch next pillar row for this lane (gfx1250 global_prefetch_b8)
        if (p + nWaves < nP) {
            __builtin_prefetch(pillars + (size_t)(p + nWaves) * T_PTS * RAW_CH +
                                   (size_t)lane * RAW_CH, 0, 2);
        }

        // lane t owns point t: 80B row, 16B aligned -> 5x b128 loads
        const float4* prow = (const float4*)(pillars +
                             (size_t)p * T_PTS * RAW_CH + (size_t)lane * RAW_CH);
        const float4 q0 = prow[0], q1 = prow[1], q2 = prow[2],
                     q3 = prow[3], q4 = prow[4];

        // xyz mean over ALL 32 points (reference sums padded points too),
        // divided by num_points
        float sx = q0.x, sy = q0.y, sz = q0.z;
#pragma unroll
        for (int off = 16; off > 0; off >>= 1) {
            sx += __shfl_xor(sx, off);
            sy += __shfl_xor(sy, off);
            sz += __shfl_xor(sz, off);
        }
        const float invn = 1.0f / (float)num_points[p];
        const float mx = sx * invn, my = sy * invn, mz = sz * invn;

        const int bb = coords[p * 4 + 0];
        const int zz = coords[p * 4 + 1];
        const int yy = coords[p * 4 + 2];
        const int xx = coords[p * 4 + 3];

        const float clx = q0.x - mx, cly = q0.y - my, clz = q0.z - mz;
        const float cex = q0.x - ((float)xx * 0.16f + 0.08f);           // X_OFF
        const float cey = q0.y - ((float)yy * 0.16f + (0.08f - 39.68f)); // Y_OFF
        const float cez = q0.z - ((float)zz * 4.0f  + (2.0f - 3.0f));    // Z_OFF

        // 26 features -> f16, pad to 32, store row t to LDS (4x ds_store_b128)
        v8h h0 = { (_Float16)q0.x, (_Float16)q0.y, (_Float16)q0.z, (_Float16)q0.w,
                   (_Float16)q1.x, (_Float16)q1.y, (_Float16)q1.z, (_Float16)q1.w };
        v8h h1 = { (_Float16)q2.x, (_Float16)q2.y, (_Float16)q2.z, (_Float16)q2.w,
                   (_Float16)q3.x, (_Float16)q3.y, (_Float16)q3.z, (_Float16)q3.w };
        v8h h2 = { (_Float16)q4.x, (_Float16)q4.y, (_Float16)q4.z, (_Float16)q4.w,
                   (_Float16)clx,  (_Float16)cly,  (_Float16)clz,  (_Float16)cex };
        v8h h3 = { (_Float16)cey,  (_Float16)cez,  (_Float16)0.f,  (_Float16)0.f,
                   (_Float16)0.f,  (_Float16)0.f,  (_Float16)0.f,  (_Float16)0.f };
        v8h* drow = (v8h*)&lds_feats[wave][lane][0];
        drow[0] = h0; drow[1] = h1; drow[2] = h2; drow[3] = h3;

        // LDS pipe is in-order per wave; wait anyway before fragment reads
        asm volatile("s_wait_dscnt 0" ::: "memory");

        // ---- A fragments (contiguous K order, same as B; not AMD's table order):
        // lanes 0-15: row = mt*16 + lane, K 0..15
        // lanes 16-31: same row, K 16..31
        v8f acc[2][4];
#pragma unroll
        for (int mt = 0; mt < 2; ++mt)
#pragma unroll
            for (int nt = 0; nt < 4; ++nt)
                acc[mt][nt] = (v8f){0.f, 0.f, 0.f, 0.f, 0.f, 0.f, 0.f, 0.f};

        const int kb = halfSel * 16;
#pragma unroll
        for (int mt = 0; mt < 2; ++mt) {
            const int row = mt * 16 + lcol;
            const v8h lo = *(const v8h*)&lds_feats[wave][row][kb];
            const v8h hi = *(const v8h*)&lds_feats[wave][row][kb + 8];
            const v16h a = __builtin_shufflevector(lo, hi, 0, 1, 2, 3, 4, 5, 6, 7,
                                                   8, 9, 10, 11, 12, 13, 14, 15);
#pragma unroll
            for (int nt = 0; nt < 4; ++nt) {
                acc[mt][nt] = __builtin_amdgcn_wmma_f32_16x16x32_f16(
                    false, a, false, bfrag[nt], (short)0, acc[mt][nt],
                    false, false);
            }
        }

        // ---- BN + ReLU + max over 32 rows.
        // C/D layout: lane = column (lcol), VGPR j = row j (+8 for lanes 16-31)
        float mcol[4];
#pragma unroll
        for (int nt = 0; nt < 4; ++nt) {
            float m = 0.0f;  // ReLU output is >= 0, so 0 is a valid identity
#pragma unroll
            for (int mt = 0; mt < 2; ++mt)
#pragma unroll
                for (int j = 0; j < 8; ++j) {
                    float v = acc[mt][nt][j] * bnScale[nt] + bnBias[nt];
                    v = fmaxf(v, 0.0f);
                    m = fmaxf(m, v);
                }
            m = fmaxf(m, __shfl_xor(m, 16));  // combine row halves
            mcol[nt] = m;                     // all 32 lanes hold column max
        }

        // ---- scatter: out[b, c, z + y*NX + x], lanes 0-15 -> tiles 0,1;
        // lanes 16-31 -> tiles 2,3 (same values after the xor-16 max)
        const size_t baseOut = (size_t)bb * C_OUTN * (size_t)NYX +
                               (size_t)zz + (size_t)yy * NXC + (size_t)xx;
        const int c0 = halfSel * 32 + lcol;
        out[baseOut + (size_t)c0 * NYX]        = mcol[halfSel * 2 + 0];
        out[baseOut + (size_t)(c0 + 16) * NYX] = mcol[halfSel * 2 + 1];
    }
}

extern "C" void kernel_launch(void* const* d_in, const int* in_sizes, int n_in,
                              void* d_out, int out_size, void* d_ws, size_t ws_size,
                              hipStream_t stream) {
    const float* pillars    = (const float*)d_in[0];
    const float* W          = (const float*)d_in[1];
    const float* gamma      = (const float*)d_in[2];
    const float* beta       = (const float*)d_in[3];
    const float* run_mean   = (const float*)d_in[4];
    const float* run_var    = (const float*)d_in[5];
    const int*   coords     = (const int*)d_in[6];
    const int*   num_points = (const int*)d_in[7];
    float* out = (float*)d_out;

    const int nP = in_sizes[7];       // P = 80000 pillars
    const int n4 = out_size / 4;      // out_size divisible by 4

    // 1) zero the 219MB output map (memory-bandwidth bound; ~9.4us floor)
    pfe_zero_kernel<<<4096, 256, 0, stream>>>((float4*)out, n4);

    // 2) PFE: 2500 blocks x 4 waves = 10000 waves, 8 pillars per wave
    const int blocks = 2500;
    pfe_wmma_kernel<<<blocks, WAVES_PER_BLOCK * 32, 0, stream>>>(
        pillars, W, gamma, beta, run_mean, run_var, coords, num_points,
        out, nP);
}
